// DENetPartDecoder_48945447305230
// MI455X (gfx1250) — matrix operand order repaired and checked by reading, not compiled, not run on hardware
//
#include <hip/hip_runtime.h>
#include <hip/hip_bf16.h>
#include <stdint.h>

// ---------------------------------------------------------------------------
// CDNA5 (gfx1250) implementation of the PointNet++ part-seg decoder.
// GEMMs (1x1 convs) via v_wmma_f32_16x16x32_bf16 (wave32, f32 accumulate),
// double-buffered LDS tiles with register-staged global loads.
// ---------------------------------------------------------------------------

typedef __attribute__((ext_vector_type(16))) __bf16 v16bf;
typedef __attribute__((ext_vector_type(8)))  float  v8f;

#define GTM 128   // block tile rows  (Cout)
#define GTN 128   // block tile cols  (M = B*N)
#define GTK 32    // K step (one bf16 WMMA)
#define LDS_H 40  // padded row length in halves (80B: 16B aligned, conflict-friendly)

// ---------------------------------------------------------------------------
// GEMM: Y(Cout x M) = W(Cout x Cin, row-major) * X(Cin x M, row-major)
// Cout % 128 == 0, M % 128 == 0, Cin % 32 == 0  (guaranteed by problem sizes)
// ---------------------------------------------------------------------------
__global__ __launch_bounds__(256) void gemm_bf16_wmma_kernel(
    const float* __restrict__ W,
    const float* __restrict__ X,
    float* __restrict__ Y,
    int Cout, int Cin, int M)
{
  __shared__ __align__(16) __bf16 Alds[2][GTM][LDS_H];
  __shared__ __align__(16) __bf16 Blds[2][GTN][LDS_H];

  const int tid  = threadIdx.x;
  const int lane = tid & 31;
  const int wave = tid >> 5;
  const int waveRow = (wave >> 1) * 32;   // 0,32,64,96
  const int waveCol = (wave & 1) * 64;    // 0,64
  const int rowBase = blockIdx.y * GTM;
  const int colBase = blockIdx.x * GTN;
  const int ml  = lane & 15;              // row (A) / col (B) within 16-tile
  const int grp = lane >> 4;              // K interleave group

  v8f acc[2][4];
  const v8f vzero = {0.f, 0.f, 0.f, 0.f, 0.f, 0.f, 0.f, 0.f};
#pragma unroll
  for (int i = 0; i < 2; ++i)
#pragma unroll
    for (int j = 0; j < 4; ++j) acc[i][j] = vzero;

  // register staging for one (A,B) tile pair: 8x float4 per thread
  float4 aR[4], bR[4];

  auto loadRegs = [&](int kt) {
    const int k0 = kt * GTK;
#pragma unroll
    for (int it = 0; it < 4; ++it) {          // A: 128 rows x 8 float4s
      int lin = tid + it * 256;
      int r   = lin >> 3;
      int f4  = lin & 7;
      aR[it] = *(const float4*)(W + (size_t)(rowBase + r) * Cin + k0 + f4 * 4);
    }
#pragma unroll
    for (int it = 0; it < 4; ++it) {          // B: 32 K rows x 32 float4s
      int lin = tid + it * 256;
      int kk  = lin >> 5;
      int n4  = lin & 31;
      bR[it] = *(const float4*)(X + (size_t)(k0 + kk) * M + colBase + n4 * 4);
    }
  };

  auto storeLds = [&](int buf) {
#pragma unroll
    for (int it = 0; it < 4; ++it) {          // A tile: K contiguous per row
      int lin = tid + it * 256;
      int r   = lin >> 3;
      int f4  = lin & 7;
      __bf16* dst = &Alds[buf][r][f4 * 4];
      dst[0] = (__bf16)aR[it].x; dst[1] = (__bf16)aR[it].y;
      dst[2] = (__bf16)aR[it].z; dst[3] = (__bf16)aR[it].w;
    }
#pragma unroll
    for (int it = 0; it < 4; ++it) {          // B tile transposed: [col][K]
      int lin = tid + it * 256;
      int kk  = lin >> 5;
      int n4  = lin & 31;
      Blds[buf][n4 * 4 + 0][kk] = (__bf16)bR[it].x;
      Blds[buf][n4 * 4 + 1][kk] = (__bf16)bR[it].y;
      Blds[buf][n4 * 4 + 2][kk] = (__bf16)bR[it].z;
      Blds[buf][n4 * 4 + 3][kk] = (__bf16)bR[it].w;
    }
  };

  const int kTiles = Cin / GTK;

  // prologue: tile 0 into buffer 0
  loadRegs(0);
  storeLds(0);
  __syncthreads();

  for (int kt = 0; kt < kTiles; ++kt) {
    const int cur = kt & 1;

    // issue next tile's global loads; they stay in flight during the WMMAs
    if (kt + 1 < kTiles) loadRegs(kt + 1);

    // ---- fragments (ISA 16-bit layouts; contiguous 16B ds_load_b128) -----
    // A 16x32: lanes g=0 hold K 0..7 & 16..23, g=1 hold K 8..15 & 24..31
    v16bf afrag[2];
#pragma unroll
    for (int i = 0; i < 2; ++i) {
      union { uint32_t u[8]; v16bf v; } fa;
      const uint32_t* q0 = (const uint32_t*)&Alds[cur][waveRow + 16 * i + ml][8 * grp];
      const uint32_t* q1 = (const uint32_t*)&Alds[cur][waveRow + 16 * i + ml][16 + 8 * grp];
      fa.u[0] = q0[0]; fa.u[1] = q0[1]; fa.u[2] = q0[2]; fa.u[3] = q0[3];
      fa.u[4] = q1[0]; fa.u[5] = q1[1]; fa.u[6] = q1[2]; fa.u[7] = q1[3];
      afrag[i] = fa.v;
    }
    // B 32x16: lanes g=0 hold K 0..7 & 16..23, g=1 hold K 8..15 & 24..31, col = lane&15
    v16bf bfrag[4];
#pragma unroll
    for (int j = 0; j < 4; ++j) {
      union { uint32_t u[8]; v16bf v; } fb;
      const uint32_t* q = (const uint32_t*)&Blds[cur][waveCol + 16 * j + ml][8 * grp];
      const uint32_t* q1 = (const uint32_t*)&Blds[cur][waveCol + 16 * j + ml][16 + 8 * grp];
#pragma unroll
      for (int h = 0; h < 4; ++h) { fb.u[h] = q[h]; fb.u[4 + h] = q1[h]; }
      bfrag[j] = fb.v;
    }

#pragma unroll
    for (int i = 0; i < 2; ++i)
#pragma unroll
      for (int j = 0; j < 4; ++j)
        acc[i][j] = __builtin_amdgcn_wmma_f32_16x16x32_bf16(
            false, afrag[i], false, bfrag[j], (short)0, acc[i][j], false, false);

    // write next tile into the other buffer (safe: everyone read it last iter
    // and the barrier below publishes it before the next iteration's reads)
    if (kt + 1 < kTiles) storeLds((kt + 1) & 1);
    __syncthreads();
  }

  // ---- epilogue: C/D layout -> row = 16*i + 8*grp + v, col = 16*j + ml ----
#pragma unroll
  for (int i = 0; i < 2; ++i)
#pragma unroll
    for (int j = 0; j < 4; ++j) {
      int col = colBase + waveCol + 16 * j + ml;
#pragma unroll
      for (int v = 0; v < 8; ++v) {
        int row = rowBase + waveRow + 16 * i + 8 * grp + v;
        Y[(size_t)row * M + col] = acc[i][j][v];
      }
    }
}

// ---------------------------------------------------------------------------
// 3-NN: for each dense point find 3 nearest sparse points + inv-dist weights
// grid: (ceil(N1/256), B); block 256.  N2 <= 2048 staged in LDS.
// ---------------------------------------------------------------------------
__global__ __launch_bounds__(256) void three_nn_kernel(
    const float* __restrict__ pd,   // (B, N1, 3)
    const float* __restrict__ ps,   // (B, N2, 3)
    int N1, int N2,
    int* __restrict__ idx3, float* __restrict__ w3)   // (B, N1, 3)
{
  __shared__ float sp[2048 * 3];
  const int b = blockIdx.y;
  for (int i = threadIdx.x; i < N2 * 3; i += 256)
    sp[i] = ps[(size_t)b * N2 * 3 + i];
  __syncthreads();

  const int n = blockIdx.x * 256 + threadIdx.x;
  if (n >= N1) return;
  const float* q = pd + ((size_t)b * N1 + n) * 3;
  const float x = q[0], y = q[1], z = q[2];

  float b0 = 3.4e38f, b1 = 3.4e38f, b2 = 3.4e38f;
  int i0 = 0, i1 = 0, i2 = 0;
  for (int j = 0; j < N2; ++j) {
    float dx = x - sp[3 * j + 0];
    float dy = y - sp[3 * j + 1];
    float dz = z - sp[3 * j + 2];
    float d = dx * dx + dy * dy + dz * dz;
    if (d < b0)      { b2 = b1; i2 = i1; b1 = b0; i1 = i0; b0 = d; i0 = j; }
    else if (d < b1) { b2 = b1; i2 = i1; b1 = d; i1 = j; }
    else if (d < b2) { b2 = d; i2 = j; }
  }
  float w0 = 1.f / (b0 + 1e-8f);
  float w1 = 1.f / (b1 + 1e-8f);
  float w2 = 1.f / (b2 + 1e-8f);
  float inv = 1.f / (w0 + w1 + w2);
  size_t o = ((size_t)b * N1 + n) * 3;
  idx3[o + 0] = i0; idx3[o + 1] = i1; idx3[o + 2] = i2;
  w3[o + 0] = w0 * inv; w3[o + 1] = w1 * inv; w3[o + 2] = w2 * inv;
}

// ---------------------------------------------------------------------------
// Build X = concat(f_skip (+cls), interp(f_sparse)) in (C_total, B*N1) layout
// f_sparse addressed as base + c2*sC + b*sB + idx  (handles both layouts)
// ---------------------------------------------------------------------------
__global__ __launch_bounds__(256) void interp_concat_kernel(
    const float* __restrict__ fskip,   // (B, Cs, N1)
    const float* __restrict__ cls,     // (B, Cs) or nullptr
    const float* __restrict__ fsp,
    int sC, int sB,
    const int* __restrict__ idx3, const float* __restrict__ w3,
    float* __restrict__ X,
    int Cs, int Csp, int B, int N1)
{
  const int t = blockIdx.x * 256 + threadIdx.x;
  const int total = (Cs + Csp) * B * N1;
  if (t >= total) return;
  const int BN = B * N1;
  const int c = t / BN;
  const int r = t - c * BN;
  const int b = r / N1;
  const int n = r - b * N1;
  float out;
  if (c < Cs) {
    out = fskip[((size_t)b * Cs + c) * N1 + n];
    if (cls) out += cls[b * Cs + c];
  } else {
    const int c2 = c - Cs;
    const float* base = fsp + (size_t)c2 * sC + (size_t)b * sB;
    const int*   ii = idx3 + (size_t)(b * N1 + n) * 3;
    const float* ww = w3   + (size_t)(b * N1 + n) * 3;
    out = ww[0] * base[ii[0]] + ww[1] * base[ii[1]] + ww[2] * base[ii[2]];
  }
  X[t] = out;
}

// ---------------------------------------------------------------------------
// BN stats: per-channel sum/sumsq over M columns -> scale/shift
// grid = C blocks, 256 threads
// ---------------------------------------------------------------------------
__global__ __launch_bounds__(256) void reduce_bn_kernel(
    const float* __restrict__ Y, int M,
    const float* __restrict__ g, const float* __restrict__ bt,
    float* __restrict__ scale, float* __restrict__ shift)
{
  __shared__ float ss[256], qq[256];
  const int c = blockIdx.x;
  const float* row = Y + (size_t)c * M;
  float s = 0.f, q = 0.f;
  for (int i = threadIdx.x; i < M; i += 256) {
    float x = row[i];
    s += x; q += x * x;
  }
  ss[threadIdx.x] = s; qq[threadIdx.x] = q;
  __syncthreads();
  for (int off = 128; off > 0; off >>= 1) {
    if (threadIdx.x < off) {
      ss[threadIdx.x] += ss[threadIdx.x + off];
      qq[threadIdx.x] += qq[threadIdx.x + off];
    }
    __syncthreads();
  }
  if (threadIdx.x == 0) {
    float m = ss[0] / (float)M;
    float v = qq[0] / (float)M - m * m;
    float sc = g[c] * rsqrtf(v + 1e-5f);
    scale[c] = sc;
    shift[c] = bt[c] - m * sc;
  }
}

__global__ __launch_bounds__(256) void bn_apply_kernel(
    float* __restrict__ Y, const float* __restrict__ scale,
    const float* __restrict__ shift, int C, int M)
{
  const int t = blockIdx.x * 256 + threadIdx.x;
  if (t >= C * M) return;
  const int c = t / M;
  Y[t] = Y[t] * scale[c] + shift[c];
}

// final BN + permute (C, B*N) -> (B, C, N) into d_out
__global__ __launch_bounds__(256) void bn_final_kernel(
    const float* __restrict__ Y, const float* __restrict__ scale,
    const float* __restrict__ shift, float* __restrict__ out,
    int C, int B, int N)
{
  const int t = blockIdx.x * 256 + threadIdx.x;
  if (t >= C * B * N) return;
  const int BN = B * N;
  const int c = t / BN;
  const int r = t - c * BN;
  const int b = r / N;
  const int n = r - b * N;
  out[((size_t)b * C + c) * N + n] = Y[t] * scale[c] + shift[c];
}

// ---------------------------------------------------------------------------
// Class-embedding path: one_hot(16) -> conv 16->64 -> BN(over batch; feature
// constant over points) -> exact GELU -> conv 64->128. Output (B,128).
// single block, 128 threads.
// ---------------------------------------------------------------------------
__global__ __launch_bounds__(128) void cls_feat_kernel(
    const int* __restrict__ label,     // (B,1)
    const float* __restrict__ Wc1,     // (64,16)
    const float* __restrict__ gc, const float* __restrict__ bc,
    const float* __restrict__ Wc2,     // (128,64)
    float* __restrict__ cls)           // (B,128)
{
  __shared__ float act[4][64];
  const int tid = threadIdx.x;
  if (tid < 64) {
    const int j = tid;
    float vals[4];
    float m = 0.f;
#pragma unroll
    for (int b = 0; b < 4; ++b) {
      vals[b] = Wc1[j * 16 + label[b]];
      m += vals[b];
    }
    m *= 0.25f;
    float v = 0.f;
#pragma unroll
    for (int b = 0; b < 4; ++b) { float d = vals[b] - m; v += d * d; }
    v *= 0.25f;
    const float inv = rsqrtf(v + 1e-5f);
#pragma unroll
    for (int b = 0; b < 4; ++b) {
      float xn = gc[j] * (vals[b] - m) * inv + bc[j];
      act[b][j] = xn * 0.5f * (1.f + erff(xn * 0.70710678118654752f));
    }
  }
  __syncthreads();
  const int c = tid;  // 0..127
#pragma unroll
  for (int b = 0; b < 4; ++b) {
    float s = 0.f;
    for (int j = 0; j < 64; ++j) s += Wc2[c * 64 + j] * act[b][j];
    cls[b * 128 + c] = s;
  }
}

// ---------------------------------------------------------------------------
extern "C" void kernel_launch(void* const* d_in, const int* in_sizes, int n_in,
                              void* d_out, int out_size, void* d_ws, size_t ws_size,
                              hipStream_t stream) {
  (void)in_sizes; (void)n_in; (void)out_size; (void)ws_size;
  const int B = 4;

  const float* p1 = (const float*)d_in[1];
  const float* p2 = (const float*)d_in[2];
  const float* p3 = (const float*)d_in[3];
  const float* p4 = (const float*)d_in[4];
  const float* f1 = (const float*)d_in[5];
  const float* f2 = (const float*)d_in[6];
  const float* f3 = (const float*)d_in[7];
  const float* f4 = (const float*)d_in[8];
  const int*   clsl = (const int*)d_in[9];
  const float* Wc1 = (const float*)d_in[10];
  const float* gc  = (const float*)d_in[11];
  const float* bc  = (const float*)d_in[12];
  const float* Wc2 = (const float*)d_in[13];
  const float* Ws2a = (const float*)d_in[14];
  const float* gs2a = (const float*)d_in[15];
  const float* bs2a = (const float*)d_in[16];
  const float* Ws2b = (const float*)d_in[17];
  const float* gs2b = (const float*)d_in[18];
  const float* bs2b = (const float*)d_in[19];
  const float* Ws1a = (const float*)d_in[20];
  const float* gs1a = (const float*)d_in[21];
  const float* bs1a = (const float*)d_in[22];
  const float* Ws1b = (const float*)d_in[23];
  const float* gs1b = (const float*)d_in[24];
  const float* bs1b = (const float*)d_in[25];
  const float* Ws0a = (const float*)d_in[26];
  const float* gs0a = (const float*)d_in[27];
  const float* bs0a = (const float*)d_in[28];
  const float* Ws0b = (const float*)d_in[29];
  const float* gs0b = (const float*)d_in[30];
  const float* bs0b = (const float*)d_in[31];

  // workspace layout (floats); regions reused across stages
  float* ws = (float*)d_ws;
  float* X  = ws;                       // 384*32768 = 12,582,912 (max concat)
  float* Y1 = X  + 12582912;            // 128*32768 =  4,194,304 (max Y)
  float* Y2 = Y1 + 4194304;             //               4,194,304 (holds f3n/f2n)
  int*   idxb = (int*)(Y2 + 4194304);   // B*8192*3  =     98,304
  float* wgtb = (float*)(idxb + 98304); //                 98,304
  float* scA  = wgtb + 98304;           // 512
  float* shA  = scA + 512;              // 512
  float* cls  = shA + 512;              // 512  (B x 128)

  // class embedding (B,128), constant over points
  cls_feat_kernel<<<1, 128, 0, stream>>>(clsl, Wc1, gc, bc, Wc2, cls);

  // ------------------- stage s2: p4(128) -> p3(512) ----------------------
  {
    const int N1 = 512, N2 = 128, Cs = 512, Csp = 1024, Co = 512, Ci = 1536;
    const int M = B * N1;
    three_nn_kernel<<<dim3((N1 + 255) / 256, B), 256, 0, stream>>>(p3, p4, N1, N2, idxb, wgtb);
    interp_concat_kernel<<<((Cs + Csp) * M + 255) / 256, 256, 0, stream>>>(
        f3, nullptr, f4, /*sC=*/N2, /*sB=*/Csp * N2, idxb, wgtb, X, Cs, Csp, B, N1);
    gemm_bf16_wmma_kernel<<<dim3(M / GTN, Co / GTM), 256, 0, stream>>>(Ws2a, X, Y1, Co, Ci, M);
    reduce_bn_kernel<<<Co, 256, 0, stream>>>(Y1, M, gs2a, bs2a, scA, shA);
    bn_apply_kernel<<<(Co * M + 255) / 256, 256, 0, stream>>>(Y1, scA, shA, Co, M);
    gemm_bf16_wmma_kernel<<<dim3(M / GTN, Co / GTM), 256, 0, stream>>>(Ws2b, Y1, Y2, Co, Co, M);
    reduce_bn_kernel<<<Co, 256, 0, stream>>>(Y2, M, gs2b, bs2b, scA, shA);
    bn_apply_kernel<<<(Co * M + 255) / 256, 256, 0, stream>>>(Y2, scA, shA, Co, M);
    // Y2 = f3n, layout (512, B*512)
  }

  // ------------------- stage s1: p3(512) -> p2(2048) ---------------------
  {
    const int N1 = 2048, N2 = 512, Cs = 256, Csp = 512, Co = 256, Ci = 768;
    const int M = B * N1;
    three_nn_kernel<<<dim3((N1 + 255) / 256, B), 256, 0, stream>>>(p2, p3, N1, N2, idxb, wgtb);
    interp_concat_kernel<<<((Cs + Csp) * M + 255) / 256, 256, 0, stream>>>(
        f2, nullptr, Y2, /*sC=*/B * N2, /*sB=*/N2, idxb, wgtb, X, Cs, Csp, B, N1);
    gemm_bf16_wmma_kernel<<<dim3(M / GTN, Co / GTM), 256, 0, stream>>>(Ws1a, X, Y1, Co, Ci, M);
    reduce_bn_kernel<<<Co, 256, 0, stream>>>(Y1, M, gs1a, bs1a, scA, shA);
    bn_apply_kernel<<<(Co * M + 255) / 256, 256, 0, stream>>>(Y1, scA, shA, Co, M);
    gemm_bf16_wmma_kernel<<<dim3(M / GTN, Co / GTM), 256, 0, stream>>>(Ws1b, Y1, Y2, Co, Co, M);
    reduce_bn_kernel<<<Co, 256, 0, stream>>>(Y2, M, gs1b, bs1b, scA, shA);
    bn_apply_kernel<<<(Co * M + 255) / 256, 256, 0, stream>>>(Y2, scA, shA, Co, M);
    // Y2 = f2n, layout (256, B*2048)
  }

  // ------------------- stage s0: p2(2048) -> p1(8192) --------------------
  {
    const int N1 = 8192, N2 = 2048, Cs = 128, Csp = 256, Co = 128, Ci = 384;
    const int M = B * N1;
    three_nn_kernel<<<dim3((N1 + 255) / 256, B), 256, 0, stream>>>(p1, p2, N1, N2, idxb, wgtb);
    interp_concat_kernel<<<((Cs + Csp) * M + 255) / 256, 256, 0, stream>>>(
        f1, cls, Y2, /*sC=*/B * N2, /*sB=*/N2, idxb, wgtb, X, Cs, Csp, B, N1);
    gemm_bf16_wmma_kernel<<<dim3(M / GTN, Co / GTM), 256, 0, stream>>>(Ws0a, X, Y1, Co, Ci, M);
    reduce_bn_kernel<<<Co, 256, 0, stream>>>(Y1, M, gs0a, bs0a, scA, shA);
    bn_apply_kernel<<<(Co * M + 255) / 256, 256, 0, stream>>>(Y1, scA, shA, Co, M);
    gemm_bf16_wmma_kernel<<<dim3(M / GTN, Co / GTM), 256, 0, stream>>>(Ws0b, Y1, Y2, Co, Co, M);
    reduce_bn_kernel<<<Co, 256, 0, stream>>>(Y2, M, gs0b, bs0b, scA, shA);
    bn_final_kernel<<<(Co * M + 255) / 256, 256, 0, stream>>>(
        Y2, scA, shA, (float*)d_out, Co, B, 8192);
  }
}
